// MambaBlock_22832046145858
// MI455X (gfx1250) — matrix operand order repaired and checked by reading, not compiled, not run on hardware
//
#include <hip/hip_runtime.h>
#include <hip/hip_bf16.h>
#include <stdint.h>

#define BB 8
#define TT 1536
#define CC 512
#define DIN 1024
#define NN 16
#define DT_RANK 32
#define KK 4
#define TCHUNK 512
#define MTOT (BB*TT)   // 12288 rows (b*T + t), 16-row tiles never straddle a chunk

typedef __attribute__((ext_vector_type(16))) __bf16 v16bf;
typedef __attribute__((ext_vector_type(8)))  __bf16 v8bf;
typedef __attribute__((ext_vector_type(8)))  float  v8f;

union FragAB { v16bf f; v8bf h[2]; unsigned short u[16]; };

__device__ __forceinline__ unsigned short f2bf(float x){
  unsigned u = __float_as_uint(x);
  u += 0x7fffu + ((u >> 16) & 1u);      // round-to-nearest-even
  return (unsigned short)(u >> 16);
}
__device__ __forceinline__ float bf2f(unsigned short s){
  return __uint_as_float(((unsigned)s) << 16);
}
// branch-free softplus: max(x,0) + log(1+exp(-|x|)) -- no EXEC-modifying regions
__device__ __forceinline__ float softplus_fast(float x){
  return fmaxf(x, 0.f) + __logf(1.f + __expf(-fabsf(x)));
}

// ---------------- f32 -> bf16 convert ----------------
__global__ void k_cvt_bf16(const float* __restrict__ src, unsigned short* __restrict__ dst, int n){
  int i = blockIdx.x * blockDim.x + threadIdx.x;
  if (i < n) dst[i] = f2bf(src[i]);
}

// ------------- pack B [3][K][N] f32 -> WMMA-B fragment-major bf16 -------------
// dst per chunk: [Ntiles][Ktiles][lane 0..31][elem 0..15]
// B frag layout (16x16x32 bf16): lane L -> n = L%16, half = L/16; elem e -> k = half*16 + e
__global__ void k_pack_b(const float* __restrict__ src, unsigned short* __restrict__ dst,
                         int Kdim, int Ndim, int Ktiles, int Ntiles){
  long total = 3L * Ntiles * Ktiles * 32 * 16;
  long i = (long)blockIdx.x * blockDim.x + threadIdx.x;
  if (i >= total) return;
  int e    = (int)(i & 15);
  int lane = (int)((i >> 4) & 31);
  long rest = i >> 9;
  int kt = (int)(rest % Ktiles); rest /= Ktiles;
  int nt = (int)(rest % Ntiles);
  int chunk = (int)(rest / Ntiles);
  int half = lane >> 4;
  int n = nt * 16 + (lane & 15);
  int k = kt * 32 + half * 16 + e;
  dst[i] = f2bf(src[(long)chunk * Kdim * Ndim + (long)k * Ndim + n]);
}

// ---------------- generic bf16 WMMA GEMM: C[M,N] = A[M,K] x B_chunk[K,N] (+residual) ----
// One wave = 16x64 tile; K loop fully unrolled (KTILES compile-time) with ping-pong
// double buffering. B addressing = uniform SGPR base + single 32-bit lane offset +
// compile-time immediate (no per-tile 64-bit vector pointers -> no spills).
// Grid exactly covers n-tile groups: (waves/block)*gridDim.x == Ntiles/4. No guards,
// EXEC stays all-1s around every WMMA.
template<int KTILES>
__global__ __launch_bounds__(128) void k_gemm(
    const unsigned short* __restrict__ A, int lda,
    const unsigned short* __restrict__ Bp, int Ntiles,
    float* __restrict__ C, int Ndim,
    const float* __restrict__ residual)
{
  const int lane  = threadIdx.x & 31;
  const int wave  = threadIdx.x >> 5;
  const int wpb   = blockDim.x >> 5;
  const int mtile = blockIdx.y;
  const int nt0   = (blockIdx.x * wpb + wave) * 4;
  const int row0  = mtile * 16;
  const int chunk = (row0 % TT) / TCHUNK;
  const int half  = lane >> 4;
  const int m     = row0 + (lane & 15);
  const unsigned short* Abase = A + (long)m * lda + half * 16; // per-lane 64-bit base
  const unsigned short* Bc    = Bp + (long)chunk * Ntiles * KTILES * 512; // uniform base
  const int bOff = nt0 * KTILES * 512 + lane * 16;            // single 32-bit vec offset

  v8f acc0 = v8f{}, acc1 = v8f{}, acc2 = v8f{}, acc3 = v8f{};
  FragAB aP, bP[4], aQ, bQ[4];

  // loads with compile-time constant displacements
  #define LOAD_A(frag, kt) do { \
      (frag).h[0] = *(const v8bf*)(Abase + (kt) * 32); \
      (frag).h[1] = *(const v8bf*)(Abase + (kt) * 32 + 8); } while (0)
  #define LOAD_B(arr, kt) do { _Pragma("unroll") \
      for (int j = 0; j < 4; ++j) { \
        (arr)[j].h[0] = *(const v8bf*)(Bc + bOff + (j * KTILES + (kt)) * 512); \
        (arr)[j].h[1] = *(const v8bf*)(Bc + bOff + (j * KTILES + (kt)) * 512 + 8); } } while (0)

  LOAD_A(aP, 0);
  LOAD_B(bP, 0);
  #pragma unroll
  for (int kt = 0; kt < KTILES; kt += 2) {
    LOAD_A(aQ, kt + 1);                      // prefetch kt+1 while computing kt
    LOAD_B(bQ, kt + 1);
    acc0 = __builtin_amdgcn_wmma_f32_16x16x32_bf16(false, aP.f, false, bP[0].f, (short)0, acc0, false, false);
    acc1 = __builtin_amdgcn_wmma_f32_16x16x32_bf16(false, aP.f, false, bP[1].f, (short)0, acc1, false, false);
    acc2 = __builtin_amdgcn_wmma_f32_16x16x32_bf16(false, aP.f, false, bP[2].f, (short)0, acc2, false, false);
    acc3 = __builtin_amdgcn_wmma_f32_16x16x32_bf16(false, aP.f, false, bP[3].f, (short)0, acc3, false, false);
    if (kt + 2 < KTILES) {                   // compile-time resolved tail
      LOAD_A(aP, kt + 2);
      LOAD_B(bP, kt + 2);
    }
    acc0 = __builtin_amdgcn_wmma_f32_16x16x32_bf16(false, aQ.f, false, bQ[0].f, (short)0, acc0, false, false);
    acc1 = __builtin_amdgcn_wmma_f32_16x16x32_bf16(false, aQ.f, false, bQ[1].f, (short)0, acc1, false, false);
    acc2 = __builtin_amdgcn_wmma_f32_16x16x32_bf16(false, aQ.f, false, bQ[2].f, (short)0, acc2, false, false);
    acc3 = __builtin_amdgcn_wmma_f32_16x16x32_bf16(false, aQ.f, false, bQ[3].f, (short)0, acc3, false, false);
  }
  #undef LOAD_A
  #undef LOAD_B

  // D layout: lane -> n = lane%16; vgpr r -> m = row0 + r + half*8
  v8f accs[4] = {acc0, acc1, acc2, acc3};
  const int ncol = (lane & 15);
  if (residual) {
    #pragma unroll
    for (int j = 0; j < 4; ++j) {
      int n0 = (nt0 + j) * 16 + ncol;
      #pragma unroll
      for (int r = 0; r < 8; ++r) {
        long idx = (long)(row0 + r + half * 8) * Ndim + n0;
        C[idx] = accs[j][r] + residual[idx];
      }
    }
  } else {
    #pragma unroll
    for (int j = 0; j < 4; ++j) {
      int n0 = (nt0 + j) * 16 + ncol;
      #pragma unroll
      for (int r = 0; r < 8; ++r) {
        long idx = (long)(row0 + r + half * 8) * Ndim + n0;
        C[idx] = accs[j][r];
      }
    }
  }
}

// ---------------- dt GEMM: dt = softplus(dbc[:, :32] @ Wdt + bdt), K=32, N=1024 --------
// Grid: gridDim.x * (blockDim.x/32) waves == Ntiles/4 == 16 exactly. No guards; softplus
// is branch-free so EXEC is never modified between the unrolled WMMAs.
__global__ __launch_bounds__(128) void k_gemm_dt(
    const float* __restrict__ dbc,           // [MTOT, 64]
    const unsigned short* __restrict__ Bp,   // Wdt packed, Ktiles=1, Ntiles=64
    const float* __restrict__ bdt,           // [3, DIN]
    float* __restrict__ dt)                  // [MTOT, DIN]
{
  const int lane  = threadIdx.x & 31;
  const int wave  = threadIdx.x >> 5;
  const int mtile = blockIdx.y;
  const int nt0   = (blockIdx.x * 4 + wave) * 4;
  const int Ntiles = 64;
  const int row0  = mtile * 16;
  const int chunk = (row0 % TT) / TCHUNK;
  const int half  = lane >> 4;
  const float* arow = dbc + (long)(row0 + (lane & 15)) * 64;

  FragAB a;
  #pragma unroll
  for (int e = 0; e < 8; ++e) a.u[e]     = f2bf(arow[half * 16 + e]);
  #pragma unroll
  for (int e = 0; e < 8; ++e) a.u[8 + e] = f2bf(arow[half * 16 + 8 + e]);

  const unsigned short* Bc = Bp + (long)chunk * Ntiles * 512;
  const int bOff = nt0 * 512 + lane * 16;
  #pragma unroll
  for (int j = 0; j < 4; ++j) {
    FragAB b;
    b.h[0] = *(const v8bf*)(Bc + bOff + j * 512);
    b.h[1] = *(const v8bf*)(Bc + bOff + j * 512 + 8);
    v8f acc = v8f{};
    acc = __builtin_amdgcn_wmma_f32_16x16x32_bf16(
        false, a.f, false, b.f, (short)0, acc, false, false);
    int n0 = (nt0 + j) * 16 + (lane & 15);
    float bias = bdt[chunk * DIN + n0];
    #pragma unroll
    for (int r = 0; r < 8; ++r) {
      dt[(long)(row0 + r + half * 8) * DIN + n0] = softplus_fast(acc[r] + bias);
    }
  }
}

// ---------------- depthwise causal conv (K=4) + SiLU, write xs as bf16 ----------------
__global__ void k_conv_silu(const float* __restrict__ xz,
                            const float* __restrict__ conv_w,   // [3, DIN, 4]
                            const float* __restrict__ conv_b,   // [3, DIN]
                            unsigned short* __restrict__ xs_bf) // [MTOT, DIN]
{
  long i = (long)blockIdx.x * blockDim.x + threadIdx.x;
  if (i >= (long)MTOT * DIN) return;
  int  d  = (int)(i % DIN);
  long g  = i / DIN;
  int  tg = (int)(g % TT);
  int  chunk = tg / TCHUNK;
  int  tc    = tg % TCHUNK;
  const float* w = conv_w + ((long)chunk * DIN + d) * KK;
  float acc = conv_b[chunk * DIN + d];
  #pragma unroll
  for (int k = 0; k < KK; ++k) {
    int ts = tc - (KK - 1) + k;
    if (ts >= 0) acc += xz[(g + (ts - tc)) * 2048 + d] * w[k];
  }
  float s = acc / (1.f + __expf(-acc));                // silu
  xs_bf[i] = f2bf(s);
}

// ---------------- selective scan + D skip + silu(z) gate --------------------------------
__global__ __launch_bounds__(256) void k_scan(
    const float* __restrict__ dt,              // [MTOT, DIN]
    const unsigned short* __restrict__ xs_bf,  // [MTOT, DIN]
    const float* __restrict__ dbc,             // [MTOT, 64]
    const float* __restrict__ xz,              // [MTOT, 2048] (z at +DIN)
    const float* __restrict__ A_log,           // [3, DIN, 16]
    const float* __restrict__ Dp,              // [3, DIN]
    unsigned short* __restrict__ y_bf)         // [MTOT, DIN]
{
  __shared__ float sB[16], sC[16];
  int blk   = blockIdx.x;                 // b*12 + chunk*4 + dgrp
  int dgrp  = blk & 3;
  int chunk = (blk >> 2) % 3;
  int b     = blk / 12;
  int d     = dgrp * 256 + threadIdx.x;

  float A[16];
  #pragma unroll
  for (int n = 0; n < 16; ++n) A[n] = -__expf(A_log[((long)chunk * DIN + d) * 16 + n]);
  float Dd = Dp[chunk * DIN + d];
  float h[16];
  #pragma unroll
  for (int n = 0; n < 16; ++n) h[n] = 0.f;

  long gbase = (long)b * TT + (long)chunk * TCHUNK;
  for (int tc = 0; tc < TCHUNK; ++tc) {
    long g = gbase + tc;
    if (threadIdx.x < 32) {
      float v = dbc[g * 64 + DT_RANK + threadIdx.x];
      if (threadIdx.x < 16) sB[threadIdx.x] = v; else sC[threadIdx.x - 16] = v;
    }
    __syncthreads();
    float dtv = dt[g * DIN + d];
    float xv  = bf2f(xs_bf[g * DIN + d]);
    float dtx = dtv * xv;
    float y = 0.f;
    #pragma unroll
    for (int n = 0; n < 16; ++n) {
      h[n] = __expf(dtv * A[n]) * h[n] + dtx * sB[n];
      y += h[n] * sC[n];
    }
    y += xv * Dd;
    float z = xz[g * 2048 + DIN + d];
    y *= z / (1.f + __expf(-z));                       // * silu(z)
    y_bf[g * DIN + d] = f2bf(y);
    __syncthreads();
  }
}

// ---------------- in-place LayerNorm over rows of d_out ---------------------------------
__global__ __launch_bounds__(256) void k_layernorm(
    float* __restrict__ out, const float* __restrict__ ln_g, const float* __restrict__ ln_b)
{
  __shared__ float red[256], red2[256];
  long row   = blockIdx.x;
  int  chunk = (int)((row % TT) / TCHUNK);
  float* r = out + row * CC;
  int t = threadIdx.x;
  float v0 = r[t], v1 = r[t + 256];
  red[t]  = v0 + v1;
  red2[t] = v0 * v0 + v1 * v1;
  __syncthreads();
  for (int s = 128; s > 0; s >>= 1) {
    if (t < s) { red[t] += red[t + s]; red2[t] += red2[t + s]; }
    __syncthreads();
  }
  float mean = red[0] * (1.f / CC);
  float var  = red2[0] * (1.f / CC) - mean * mean;
  float inv  = rsqrtf(var + 1e-5f);
  r[t]       = (v0 - mean) * inv * ln_g[chunk * CC + t]       + ln_b[chunk * CC + t];
  r[t + 256] = (v1 - mean) * inv * ln_g[chunk * CC + t + 256] + ln_b[chunk * CC + t + 256];
}

extern "C" void kernel_launch(void* const* d_in, const int* in_sizes, int n_in,
                              void* d_out, int out_size, void* d_ws, size_t ws_size,
                              hipStream_t stream) {
  const float* x      = (const float*)d_in[0];
  const float* Win    = (const float*)d_in[1];
  const float* conv_w = (const float*)d_in[2];
  const float* conv_b = (const float*)d_in[3];
  const float* Wx     = (const float*)d_in[4];
  const float* Wdt    = (const float*)d_in[5];
  const float* bdt    = (const float*)d_in[6];
  const float* A_log  = (const float*)d_in[7];
  const float* Dp     = (const float*)d_in[8];
  const float* Wout   = (const float*)d_in[9];
  const float* ln_g   = (const float*)d_in[10];
  const float* ln_b   = (const float*)d_in[11];
  float* out = (float*)d_out;

  char* wp = (char*)d_ws;
  auto carve = [&](size_t bytes) -> void* {
    void* p = (void*)wp;
    wp += (bytes + 255) & ~(size_t)255;
    return p;
  };
  unsigned short* x_bf   = (unsigned short*)carve((size_t)MTOT * CC * 2);
  unsigned short* Win_p  = (unsigned short*)carve(3ULL * 128 * 16 * 512 * 2); // N=2048,K=512
  unsigned short* Wx_p   = (unsigned short*)carve(3ULL * 4   * 32 * 512 * 2); // N=64,K=1024
  unsigned short* Wdt_p  = (unsigned short*)carve(3ULL * 64  * 1  * 512 * 2); // N=1024,K=32
  unsigned short* Wout_p = (unsigned short*)carve(3ULL * 32  * 32 * 512 * 2); // N=512,K=1024
  float*          xz     = (float*)carve((size_t)MTOT * 2048 * 4);
  unsigned short* xs_bf  = (unsigned short*)carve((size_t)MTOT * DIN * 2);
  float*          dbc    = (float*)carve((size_t)MTOT * 64 * 4);
  float*          dt     = (float*)carve((size_t)MTOT * DIN * 4);
  unsigned short* y_bf   = (unsigned short*)carve((size_t)MTOT * DIN * 2);

  // 1) convert x to bf16
  {
    int n = MTOT * CC;
    k_cvt_bf16<<<(n + 255) / 256, 256, 0, stream>>>(x, x_bf, n);
  }
  // 2) pack weights into WMMA B-fragment layout
  {
    long t1 = 3L * 128 * 16 * 512;
    k_pack_b<<<(int)((t1 + 255) / 256), 256, 0, stream>>>(Win,  Win_p,  512,  2048, 16, 128);
    long t2 = 3L * 4 * 32 * 512;
    k_pack_b<<<(int)((t2 + 255) / 256), 256, 0, stream>>>(Wx,   Wx_p,   1024, 64,   32, 4);
    long t3 = 3L * 64 * 1 * 512;
    k_pack_b<<<(int)((t3 + 255) / 256), 256, 0, stream>>>(Wdt,  Wdt_p,  32,   1024, 1,  64);
    long t4 = 3L * 32 * 32 * 512;
    k_pack_b<<<(int)((t4 + 255) / 256), 256, 0, stream>>>(Wout, Wout_p, 1024, 512,  32, 32);
  }
  const int Mtiles = MTOT / 16;  // 768
  // 3) xz = x @ Win[chunk]   (M=12288, K=512, N=2048): Ntiles=128 -> 32 groups = 8 blk x 4 waves
  k_gemm<16><<<dim3(8, Mtiles), 128, 0, stream>>>(x_bf, 512, Win_p, 128, xz, 2048, nullptr);
  // 4) depthwise conv + silu -> xs (bf16)
  {
    long n = (long)MTOT * DIN;
    k_conv_silu<<<(int)((n + 255) / 256), 256, 0, stream>>>(xz, conv_w, conv_b, xs_bf);
  }
  // 5) dbc = xs @ Wx[chunk]  (K=1024, N=64): Ntiles=4 -> 1 group = 1 blk x 1 wave (block=32)
  k_gemm<32><<<dim3(1, Mtiles), 32, 0, stream>>>(xs_bf, 1024, Wx_p, 4, dbc, 64, nullptr);
  // 6) dt = softplus(dbc[:, :32] @ Wdt[chunk] + bdt)  (K=32, N=1024): 16 groups = 4 blk x 4 waves
  k_gemm_dt<<<dim3(4, Mtiles), 128, 0, stream>>>(dbc, Wdt_p, bdt, dt);
  // 7) selective scan + skip + gate -> y (bf16)
  k_scan<<<dim3(BB * 3 * (DIN / 256)), 256, 0, stream>>>(dt, xs_bf, dbc, xz, A_log, Dp, y_bf);
  // 8) out = x + y @ Wout[chunk]  (K=1024, N=512): Ntiles=32 -> 8 groups = 2 blk x 4 waves
  k_gemm<32><<<dim3(2, Mtiles), 128, 0, stream>>>(y_bf, 1024, Wout_p, 32, out, 512, x);
  // 9) in-place LayerNorm on out rows
  k_layernorm<<<dim3(MTOT), 256, 0, stream>>>(out, ln_g, ln_b);
}
